// MultiheadAttention_22531398435337
// MI455X (gfx1250) — matrix operand order repaired and checked by reading, not compiled, not run on hardware
//
#include <hip/hip_runtime.h>

// MI455X (gfx1250) multihead attention forward.
// f16 WMMA (v_wmma_f32_16x16x32_f16) everywhere; weights pre-converted to f16
// once and staged into LDS via CDNA5 async global->LDS copies when available;
// flash-style online-softmax attention with direct-from-global Q/K fragments.
//
// Workspace layout (needs 40 MB):
//   Qh | Kh | Vh | Oh (4096*1024 f16 each) | Wqh | Wkh | Wvh | Woh (1024^2 f16)

#define HEADS    16
#define HEAD_DIM 64
#define MODEL    1024
#define SEQ      2048
#define BATCH    2
#define ROWS     (SEQ * BATCH)   // 4096 GEMM rows
#define RSTRIDE  (BATCH * MODEL) // 2048: row stride (in elements) of [S,B,M]

typedef __attribute__((ext_vector_type(16))) _Float16 v16h;
typedef __attribute__((ext_vector_type(8)))  _Float16 v8h;
typedef __attribute__((ext_vector_type(8)))  float    v8f;
typedef __attribute__((ext_vector_type(4)))  float    v4f;
typedef __attribute__((ext_vector_type(4)))  int      v4i;

// ---- CDNA5 async global->LDS copy path (ASYNCcnt-tracked, bypasses VGPRs) --
#if defined(__AMDGCN__) && __has_builtin(__builtin_amdgcn_global_load_async_to_lds_b128)
#define HAVE_ASYNC 1
#else
#define HAVE_ASYNC 0
#endif

#if HAVE_ASYNC
__device__ __forceinline__ void async_ld_b128(const void* g, void* l) {
  __builtin_amdgcn_global_load_async_to_lds_b128(
      (__attribute__((address_space(1))) v4i*)(v4i*)g,
      (__attribute__((address_space(3))) v4i*)(v4i*)l, 0, 0);
}
__device__ __forceinline__ void wait_async0() {
#if __has_builtin(__builtin_amdgcn_s_wait_asynccnt)
  __builtin_amdgcn_s_wait_asynccnt(0);
#else
  asm volatile("s_wait_asynccnt 0x0" ::: "memory");
#endif
}
#endif

__device__ __forceinline__ v8f wmma32(v16h a, v16h b, v8f c) {
  // D = A(16x32 f16) * B(32x16 f16) + C(16x16 f32)
  return __builtin_amdgcn_wmma_f32_16x16x32_f16(
      /*neg_a=*/false, a, /*neg_b=*/false, b,
      /*c_mod=*/(short)0, c, /*reuse_a=*/false, /*reuse_b=*/false);
}

// A-matrix fragment (16x32, row-major source [m][k], stride in halves).
// ISA layout: lane&15 = row m; lanes<16 hold K {0..7,16..23}, lanes>=16 hold
// K {8..15,24..31}.  Two 16B loads per lane.
__device__ __forceinline__ v16h load_frag_a(const _Float16* base, int stride_h,
                                            int lane) {
  const _Float16* p = base + (lane & 15) * stride_h + ((lane >> 4) & 1) * 8;
  v8h lo = *(const v8h*)p;
  v8h hi = *(const v8h*)(p + 16);
  v16h f;
#pragma unroll
  for (int i = 0; i < 8; ++i) { f[i] = lo[i]; f[i + 8] = hi[i]; }
  return f;
}

// B-matrix fragment (32x16), source stored as B^T row-major: [n][k].
// ISA layout: lane&15 = column n; lanes<16 hold K {0..7,16..23},
// lanes>=16 hold K {8..15,24..31} (matches A).  Two 16B loads per lane.
__device__ __forceinline__ v16h load_frag_b(const _Float16* base, int stride_h,
                                            int lane) {
  const _Float16* p = base + (lane & 15) * stride_h + ((lane >> 4) & 1) * 8;
  v8h lo = *(const v8h*)p;
  v8h hi = *(const v8h*)(p + 16);
  v16h f;
#pragma unroll
  for (int i = 0; i < 8; ++i) { f[i] = lo[i]; f[i + 8] = hi[i]; }
  return f;
}

// ---------------------------------------------------------------------------
// One-shot fp32 -> f16 conversion for weight matrices (1024x1024 each).
// ---------------------------------------------------------------------------
__global__ __launch_bounds__(256) void cvt_w_f16(const float* __restrict__ s,
                                                 _Float16* __restrict__ d) {
  const size_t i = ((size_t)blockIdx.x * 256 + threadIdx.x) * 8;
  v4f a = *(const v4f*)(s + i), b = *(const v4f*)(s + i + 4);
  v8h h;
#pragma unroll
  for (int j = 0; j < 4; ++j) { h[j] = (_Float16)a[j]; h[j + 4] = (_Float16)b[j]; }
  *(v8h*)(d + i) = h;
}

// ---------------------------------------------------------------------------
// Tiled GEMM: C[ROWS][MODEL] = A[ROWS][MODEL] * Wh[MODEL][MODEL]^T + bias
// (torch Linear convention: y[m][n] = sum_k A[m][k] * W[n][k] + b[n])
// AT = float (raw inputs, cvt fused into staging) or _Float16 (async-staged).
// Block: 256 threads (8 waves), tile 128x128, K-step 32.
// ---------------------------------------------------------------------------
template <typename AT, typename OT>
__global__ __launch_bounds__(256) void gemm_xwt(const AT* __restrict__ A,
                                                const _Float16* __restrict__ Wh,
                                                const float* __restrict__ bias,
                                                OT* __restrict__ C) {
  constexpr int K = MODEL, N = MODEL;
  constexpr int LDT = 40; // 32 + 8 halves pad (80B rows: 16B aligned)
  __shared__ __attribute__((aligned(16))) _Float16 As[128 * LDT];
  __shared__ __attribute__((aligned(16))) _Float16 Ws[128 * LDT];

  const int t = threadIdx.x, lane = t & 31, w = t >> 5;
  const int wm = w >> 2, wn = w & 3;            // 2x4 wave grid
  const int m0 = blockIdx.y * 128, n0 = blockIdx.x * 128;
  const int nl = lane & 15, half = (lane >> 4) & 1;

  v8f acc[4][2];
#pragma unroll
  for (int mt = 0; mt < 4; ++mt)
#pragma unroll
    for (int nt = 0; nt < 2; ++nt) acc[mt][nt] = (v8f){};

  for (int k0 = 0; k0 < K; k0 += 32) {
    __syncthreads();
    // Stage A and W tiles: 512 16B chunks per matrix, 2 per thread.
#pragma unroll
    for (int i = 0; i < 2; ++i) {
      const int c = t + 256 * i;
      const int r = c >> 2, kc = (c & 3) * 8;
      // W tile: f16 row-major [n][k] -> async LDS copy when available.
#if HAVE_ASYNC
      async_ld_b128(Wh + (size_t)(n0 + r) * K + k0 + kc, &Ws[r * LDT + kc]);
#else
      *(v8h*)&Ws[r * LDT + kc] =
          *(const v8h*)(Wh + (size_t)(n0 + r) * K + k0 + kc);
#endif
      // A tile.
      if constexpr (sizeof(AT) == 4) {
        const float* g = (const float*)A + (size_t)(m0 + r) * K + k0 + kc;
        v4f x0 = *(const v4f*)g, x1 = *(const v4f*)(g + 4);
        v8h ha;
#pragma unroll
        for (int j = 0; j < 4; ++j) {
          ha[j] = (_Float16)x0[j];
          ha[j + 4] = (_Float16)x1[j];
        }
        *(v8h*)&As[r * LDT + kc] = ha;
      } else {
#if HAVE_ASYNC
        async_ld_b128((const _Float16*)A + (size_t)(m0 + r) * K + k0 + kc,
                      &As[r * LDT + kc]);
#else
        *(v8h*)&As[r * LDT + kc] =
            *(const v8h*)((const _Float16*)A + (size_t)(m0 + r) * K + k0 + kc);
#endif
      }
    }
#if HAVE_ASYNC
    wait_async0();
#endif
    __syncthreads();

    v16h bf[2];
#pragma unroll
    for (int nt = 0; nt < 2; ++nt)
      bf[nt] = load_frag_b(&Ws[(wn * 32 + nt * 16) * LDT], LDT, lane);
#pragma unroll
    for (int mt = 0; mt < 4; ++mt) {
      v16h af = load_frag_a(&As[(wm * 64 + mt * 16) * LDT], LDT, lane);
#pragma unroll
      for (int nt = 0; nt < 2; ++nt)
        acc[mt][nt] = wmma32(af, bf[nt], acc[mt][nt]);
    }
  }

  // Writeout: C layout => lane holds column n; VGPR j holds row (half*8+j).
#pragma unroll
  for (int mt = 0; mt < 4; ++mt) {
#pragma unroll
    for (int nt = 0; nt < 2; ++nt) {
      const int col = n0 + wn * 32 + nt * 16 + nl;
      const float bn = bias[col];
#pragma unroll
      for (int j = 0; j < 8; ++j) {
        const int row = m0 + wm * 64 + mt * 16 + half * 8 + j;
        C[(size_t)row * N + col] = (OT)(acc[mt][nt][j] + bn);
      }
    }
  }
}

// ---------------------------------------------------------------------------
// Flash attention: O = softmax(Q K^T / sqrt(D)) V   per (batch, head)
// Grid: (SEQ/64) * BATCH * HEADS = 1024 blocks, 128 threads (4 waves).
// Each wave owns 16 query rows; key dimension streamed in blocks of 64.
// Q/K fragments load straight from global ([S,B,H,D] is already the layout
// WMMA wants); V staged transposed in LDS; P relaid via LDS.
// ---------------------------------------------------------------------------
__global__ __launch_bounds__(128) void attn_flash(const _Float16* __restrict__ Q,
                                                  const _Float16* __restrict__ Km,
                                                  const _Float16* __restrict__ V,
                                                  _Float16* __restrict__ O) {
  constexpr int LDV = 72; // V^T rows (64 keys) pad
  constexpr int LDP = 24; // P^T rows (16 m) pad
  __shared__ __attribute__((aligned(16))) _Float16 Vt[64 * LDV];    // [d][key]
  __shared__ __attribute__((aligned(16))) _Float16 Pt[4][64 * LDP]; // [key][m]

  const int t = threadIdx.x, lane = t & 31, w = t >> 5;
  const int qb = blockIdx.x & 31;   // query block (64 rows)
  const int bh = blockIdx.x >> 5;   // 0..31 : (batch, head)
  const int boff = (bh & 1) * MODEL + (bh >> 1) * HEAD_DIM;
  const int s0 = qb * 64 + w * 16;  // first query row of this wave
  const int nl = lane & 15, half = (lane >> 4) & 1;
  const float scale = 0.125f;       // 1/sqrt(64)
  _Float16* Ptw = Pt[w];

  // Preload this wave's Q fragments (16 x 64 = two 16x32 A-frags).
  v16h qf[2];
#pragma unroll
  for (int k0 = 0; k0 < 2; ++k0)
    qf[k0] = load_frag_a(Q + (size_t)s0 * RSTRIDE + boff + k0 * 32, RSTRIDE, lane);

  v8f oacc[4];
#pragma unroll
  for (int dn = 0; dn < 4; ++dn) oacc[dn] = (v8f){};
  float mprev[8], lsum[8];
#pragma unroll
  for (int j = 0; j < 8; ++j) { mprev[j] = -1e30f; lsum[j] = 0.0f; }

  for (int j0 = 0; j0 < SEQ; j0 += 64) {
    // Prefetch next key block's K and V rows into cache (uniform branch).
    if (j0 + 64 < SEQ) {
      const int pr = j0 + 64 + (t >> 1);
      const int pc = (t & 1) * 32;
      __builtin_prefetch(Km + (size_t)pr * RSTRIDE + boff + pc, 0, 3);
      __builtin_prefetch(V + (size_t)pr * RSTRIDE + boff + pc, 0, 3);
    }

    __syncthreads(); // previous block's Vt consumption complete
    // Stage V^T (64 keys x 64 d -> Vt[d][key]); 4 chunks/thread.
#pragma unroll
    for (int i = 0; i < 4; ++i) {
      const int c = t + 128 * i;
      const int key = c >> 3, d0 = (c & 7) * 8;
      v8h vv = *(const v8h*)(V + (size_t)(j0 + key) * RSTRIDE + boff + d0);
#pragma unroll
      for (int j = 0; j < 8; ++j) Vt[(d0 + j) * LDV + key] = vv[j];
    }

    // Scores S = Q K^T for 16 q-rows x 64 keys (K read straight from global).
    v8f sacc[4];
#pragma unroll
    for (int nt = 0; nt < 4; ++nt) sacc[nt] = (v8f){};
#pragma unroll
    for (int k0 = 0; k0 < 2; ++k0) {
#pragma unroll
      for (int nt = 0; nt < 4; ++nt) {
        v16h kf = load_frag_b(
            Km + (size_t)(j0 + nt * 16) * RSTRIDE + boff + k0 * 32, RSTRIDE, lane);
        sacc[nt] = wmma32(qf[k0], kf, sacc[nt]);
      }
    }

    // Online softmax. Row m = half*8+j lives across 16 lanes (columns).
    float al[8];
#pragma unroll
    for (int j = 0; j < 8; ++j) {
      float vmax = sacc[0][j];
#pragma unroll
      for (int nt = 1; nt < 4; ++nt) vmax = fmaxf(vmax, sacc[nt][j]);
      vmax *= scale;
#pragma unroll
      for (int off = 1; off < 16; off <<= 1)
        vmax = fmaxf(vmax, __shfl_xor(vmax, off, 32));
      const float mn = fmaxf(mprev[j], vmax);
      al[j] = __expf(mprev[j] - mn);
      mprev[j] = mn;
      lsum[j] *= al[j];
    }
#pragma unroll
    for (int dn = 0; dn < 4; ++dn)
#pragma unroll
      for (int j = 0; j < 8; ++j) oacc[dn][j] *= al[j];

    // P = exp(S*scale - m); store to Pt[key][m] (b128 per n-tile per lane).
#pragma unroll
    for (int nt = 0; nt < 4; ++nt) {
      v8h ph;
#pragma unroll
      for (int j = 0; j < 8; ++j) {
        const float e = __expf(sacc[nt][j] * scale - mprev[j]);
        lsum[j] += e;                     // per-lane partial; reduced at end
        ph[j] = (_Float16)e;
      }
      *(v8h*)&Ptw[(nt * 16 + nl) * LDP + half * 8] = ph;
    }
    __syncthreads(); // Vt fully staged by all waves

    // O += P V : A-frag of P gathered from Pt, B-frags from Vt.
#pragma unroll
    for (int k0 = 0; k0 < 2; ++k0) {
      v16h pf;
#pragma unroll
      for (int i = 0; i < 8; ++i) {
        pf[i]     = Ptw[(k0 * 32 + half * 8 + i) * LDP + nl];
        pf[i + 8] = Ptw[(k0 * 32 + 16 + half * 8 + i) * LDP + nl];
      }
#pragma unroll
      for (int dn = 0; dn < 4; ++dn) {
        v16h vf = load_frag_b(&Vt[(dn * 16) * LDV + k0 * 32], LDV, lane);
        oacc[dn] = wmma32(pf, vf, oacc[dn]);
      }
    }
  }

  // Finalize: full row sums, normalize, repack through LDS, b128 store.
#pragma unroll
  for (int j = 0; j < 8; ++j) {
    float l = lsum[j];
#pragma unroll
    for (int off = 1; off < 16; off <<= 1) l += __shfl_xor(l, off, 32);
    lsum[j] = 1.0f / l;
  }
#pragma unroll
  for (int dn = 0; dn < 4; ++dn)
#pragma unroll
    for (int j = 0; j < 8; ++j)
      Ptw[(half * 8 + j) * 64 + dn * 16 + nl] = (_Float16)(oacc[dn][j] * lsum[j]);
#pragma unroll
  for (int i = 0; i < 4; ++i) {
    const int c = lane + 32 * i;
    const int r = c >> 3, d0 = (c & 7) * 8;
    v8h ov = *(const v8h*)&Ptw[r * 64 + d0];
    *(v8h*)(O + (size_t)(s0 + r) * RSTRIDE + boff + d0) = ov;
  }
}

// ---------------------------------------------------------------------------
extern "C" void kernel_launch(void* const* d_in, const int* in_sizes, int n_in,
                              void* d_out, int out_size, void* d_ws,
                              size_t ws_size, hipStream_t stream) {
  (void)in_sizes; (void)n_in; (void)out_size; (void)ws_size;
  const float* q  = (const float*)d_in[0];
  const float* k  = (const float*)d_in[1];
  const float* v  = (const float*)d_in[2];
  // d_in[3] = mask: no effect in reference (applied after softmax, discarded)
  const float* Wq = (const float*)d_in[4];
  const float* bq = (const float*)d_in[5];
  const float* Wk = (const float*)d_in[6];
  const float* bk = (const float*)d_in[7];
  const float* Wv = (const float*)d_in[8];
  const float* bv = (const float*)d_in[9];
  const float* Wo = (const float*)d_in[10];
  const float* bo = (const float*)d_in[11];
  float* out = (float*)d_out;

  const size_t nel = (size_t)ROWS * MODEL;   // 4M f16 each
  const size_t wel = (size_t)MODEL * MODEL;  // 1M f16 each
  _Float16* Qh  = (_Float16*)d_ws;
  _Float16* Kh  = Qh + nel;
  _Float16* Vh  = Kh + nel;
  _Float16* Oh  = Vh + nel;
  _Float16* Wqh = Oh + nel;
  _Float16* Wkh = Wqh + wel;
  _Float16* Wvh = Wkh + wel;
  _Float16* Woh = Wvh + wel;

  // One-shot weight conversions (f32 -> f16), L2-resident thereafter.
  const int cvtg = (int)(wel / (256 * 8)); // 512 blocks
  cvt_w_f16<<<cvtg, 256, 0, stream>>>(Wq, Wqh);
  cvt_w_f16<<<cvtg, 256, 0, stream>>>(Wk, Wkh);
  cvt_w_f16<<<cvtg, 256, 0, stream>>>(Wv, Wvh);
  cvt_w_f16<<<cvtg, 256, 0, stream>>>(Wo, Woh);

  dim3 gg(MODEL / 128, ROWS / 128); // 8 x 32 tiles
  gemm_xwt<float, _Float16><<<gg, 256, 0, stream>>>(q, Wqh, bq, Qh);
  gemm_xwt<float, _Float16><<<gg, 256, 0, stream>>>(k, Wkh, bk, Kh);
  gemm_xwt<float, _Float16><<<gg, 256, 0, stream>>>(v, Wvh, bv, Vh);

  attn_flash<<<dim3((SEQ / 64) * BATCH * HEADS), 128, 0, stream>>>(Qh, Kh, Vh, Oh);

  gemm_xwt<_Float16, float><<<gg, 256, 0, stream>>>(Oh, Woh, bo, out);
}
